// EuclideanCodebook_49271864820017
// MI455X (gfx1250) — matrix operand order repaired and checked by reading, not compiled, not run on hardware
//
#include <hip/hip_runtime.h>
#include <hip/hip_bf16.h>

// Problem constants (from the reference): B=16, T=4096, C=128, K=1024
#define BT      65536
#define CH      128
#define KCODES  1024

typedef __attribute__((ext_vector_type(16))) __bf16 v16bf;
typedef __attribute__((ext_vector_type(8)))  float  v8f;

// ---------------------------------------------------------------------------
// Prep: e_sq[k] = sum_c embed[k][c]^2 (fp32), and bf16 hi/lo split of embed.
// One wave (32 lanes) per code row; each lane handles 4 channels.
// ---------------------------------------------------------------------------
__launch_bounds__(32)
__global__ void vq_prep(const float* __restrict__ embed,
                        float* __restrict__ e_sq,
                        __bf16* __restrict__ ehi,
                        __bf16* __restrict__ elo) {
  const int code = blockIdx.x;
  const int lane = threadIdx.x;
  const float* row = embed + code * CH;
  float s = 0.0f;
#pragma unroll
  for (int j = 0; j < 4; ++j) {
    const int ch = lane + 32 * j;
    const float v = row[ch];
    s += v * v;
    const __bf16 h = (__bf16)v;                    // RNE to bf16
    ehi[code * CH + ch] = h;
    elo[code * CH + ch] = (__bf16)(v - (float)h);  // residual
  }
#pragma unroll
  for (int m = 16; m >= 1; m >>= 1) s += __shfl_xor(s, m, 32);
  if (lane == 0) e_sq[code] = s;
}

// ---------------------------------------------------------------------------
// Main: per-wave 32-row block (two 16-row A slabs) x all 1024 codes.
// score = 2*(x . e) - ||e||^2  (||x||^2 dropped: argmax-invariant).
// Dot products via 3-term bf16 split on v_wmma_f32_16x16x32_bf16.
// M=32 per wave amortizes each 8KB B tile over 24 WMMAs (48 FLOP/cache byte).
// ---------------------------------------------------------------------------
__launch_bounds__(256)
__global__ void vq_argmax(const float* __restrict__ x,
                          const float* __restrict__ embed,
                          const float* __restrict__ e_sq,
                          const __bf16* __restrict__ ehi,
                          const __bf16* __restrict__ elo,
                          float* __restrict__ q_out,
                          int* __restrict__ ind_out) {
  __shared__ int s_idx[256];

  const int wave = threadIdx.x >> 5;
  const int lane = threadIdx.x & 31;
  const int sel  = lane >> 4;       // 0: lanes 0-15, 1: lanes 16-31
  const int l15  = lane & 15;
  const int m0   = blockIdx.x * 256 + wave * 32;   // first row of this wave

  // ---- Load + convert A fragments once (reused for all 64 N-tiles) ----
  // A k order used here (contiguous, not AMD's table order): lanes 0-15 hold K 0..15 of row l15,
  // lanes 16-31 hold K 16..31 of row l15.
  v16bf xh[2][4], xl[2][4];
#pragma unroll
  for (int s = 0; s < 2; ++s) {
    const float* xr = x + (size_t)(m0 + 16 * s + l15) * CH;
#pragma unroll
    for (int kc = 0; kc < 4; ++kc) {
      const int k0 = kc * 32;
      const float4* p0 = (const float4*)(xr + k0 + 16 * sel);        // K j=0..7
      const float4* p1 = (const float4*)(xr + k0 + 8 + 16 * sel);   // K j=8..15
      float va[16];
      float4 a0 = p0[0], a1 = p0[1], a2 = p1[0], a3 = p1[1];
      va[0]=a0.x;  va[1]=a0.y;  va[2]=a0.z;  va[3]=a0.w;
      va[4]=a1.x;  va[5]=a1.y;  va[6]=a1.z;  va[7]=a1.w;
      va[8]=a2.x;  va[9]=a2.y;  va[10]=a2.z; va[11]=a2.w;
      va[12]=a3.x; va[13]=a3.y; va[14]=a3.z; va[15]=a3.w;
#pragma unroll
      for (int j = 0; j < 16; ++j) {
        const __bf16 h = (__bf16)va[j];
        xh[s][kc][j] = h;
        xl[s][kc][j] = (__bf16)(va[j] - (float)h);
      }
    }
  }

  // ---- Running argmax over 64 code tiles ----
  float bv[2][8];
  int   bi[2][8];
#pragma unroll
  for (int s = 0; s < 2; ++s)
#pragma unroll
    for (int r = 0; r < 8; ++r) { bv[s][r] = -3.4e38f; bi[s][r] = 0; }

  for (int nt = 0; nt < 64; ++nt) {
    const int code = nt * 16 + l15;           // B-matrix column for this lane
    const float esq = e_sq[code];

    // Prefetch next tile of B data (one 128B line per lane covers each tile).
    if (nt + 1 < 64) {
      __builtin_prefetch((const char*)(ehi + (nt + 1) * 16 * CH) + lane * 128, 0, 1);
      __builtin_prefetch((const char*)(elo + (nt + 1) * 16 * CH) + lane * 128, 0, 1);
    }

    v8f acc0 = {};
    v8f acc1 = {};
    // B k order used here (contiguous, same as A): lane holds a contiguous run of 16 channels of
    // its code column: lanes 0-15 -> K 0..15, lanes 16-31 -> K 16..31.
    const __bf16* bh_base = ehi + (size_t)code * CH + 16 * sel;
    const __bf16* bl_base = elo + (size_t)code * CH + 16 * sel;
#pragma unroll
    for (int kc = 0; kc < 4; ++kc) {
      v16bf bh = *(const v16bf*)(bh_base + kc * 32);
      v16bf bl = *(const v16bf*)(bl_base + kc * 32);
      acc0 = __builtin_amdgcn_wmma_f32_16x16x32_bf16(false, xh[0][kc], false, bh,
                                                     (short)0, acc0, false, false);
      acc1 = __builtin_amdgcn_wmma_f32_16x16x32_bf16(false, xh[1][kc], false, bh,
                                                     (short)0, acc1, false, false);
      acc0 = __builtin_amdgcn_wmma_f32_16x16x32_bf16(false, xl[0][kc], false, bh,
                                                     (short)0, acc0, false, false);
      acc1 = __builtin_amdgcn_wmma_f32_16x16x32_bf16(false, xl[1][kc], false, bh,
                                                     (short)0, acc1, false, false);
      acc0 = __builtin_amdgcn_wmma_f32_16x16x32_bf16(false, xh[0][kc], false, bl,
                                                     (short)0, acc0, false, false);
      acc1 = __builtin_amdgcn_wmma_f32_16x16x32_bf16(false, xh[1][kc], false, bl,
                                                     (short)0, acc1, false, false);
    }

    // C/D layout: VGPR r, lanes 0-15 = row r, lanes 16-31 = row r+8; col = l15.
#pragma unroll
    for (int r = 0; r < 8; ++r) {
      const float s0 = 2.0f * acc0[r] - esq;
      if (s0 > bv[0][r]) { bv[0][r] = s0; bi[0][r] = code; }  // strict >: first max
      const float s1 = 2.0f * acc1[r] - esq;
      if (s1 > bv[1][r]) { bv[1][r] = s1; bi[1][r] = code; }
    }
  }

  // ---- Cross-lane (16-wide) max reduction with first-index tie-break ----
#pragma unroll
  for (int s = 0; s < 2; ++s) {
    const int base = m0 + 16 * s;
#pragma unroll
    for (int r = 0; r < 8; ++r) {
      float v = bv[s][r];
      int   i = bi[s][r];
#pragma unroll
      for (int m = 8; m >= 1; m >>= 1) {
        const float ov = __shfl_xor(v, m, 32);
        const int   oi = __shfl_xor(i, m, 32);
        if (ov > v || (ov == v && oi < i)) { v = ov; i = oi; }
      }
      if (lane == 0) {
        ind_out[base + r] = i;
        s_idx[wave * 32 + 16 * s + r] = i;
      }
      if (lane == 16) {
        ind_out[base + 8 + r] = i;
        s_idx[wave * 32 + 16 * s + 8 + r] = i;
      }
    }
  }
  __syncthreads();

  // ---- Gather: one thread per row copies 128 fp32 from the original embed ----
  const int tid  = threadIdx.x;          // 0..255 == local row
  const int grow = blockIdx.x * 256 + tid;
  const float4* src = (const float4*)(embed + (size_t)s_idx[tid] * CH);
  float4* dst       = (float4*)(q_out + (size_t)grow * CH);
#pragma unroll
  for (int j = 0; j < 32; ++j) dst[j] = src[j];
}

// ---------------------------------------------------------------------------
extern "C" void kernel_launch(void* const* d_in, const int* in_sizes, int n_in,
                              void* d_out, int out_size, void* d_ws, size_t ws_size,
                              hipStream_t stream) {
  const float* x     = (const float*)d_in[0];   // [16,4096,128] fp32
  const float* embed = (const float*)d_in[1];   // [1024,128]    fp32

  // Workspace layout: e_sq (4KB) | embed_hi bf16 (256KB) | embed_lo bf16 (256KB)
  float*  e_sq = (float*)d_ws;
  __bf16* ehi  = (__bf16*)((char*)d_ws + 4096);
  __bf16* elo  = (__bf16*)((char*)d_ws + 4096 + (size_t)KCODES * CH * sizeof(__bf16));

  // Output: quantized [BT*CH] fp32, then indices [BT] int32 (bit-cast).
  float* q_out   = (float*)d_out;
  int*   ind_out = (int*)(q_out + (size_t)BT * CH);

  vq_prep<<<KCODES, 32, 0, stream>>>(embed, e_sq, ehi, elo);
  vq_argmax<<<BT / 256, 256, 0, stream>>>(x, embed, e_sq, ehi, elo, q_out, ind_out);
}
